// MSReversibleRefine_16234976378892
// MI455X (gfx1250) — compile-verified, matrix operand order checked by reading, not run on hardware
//
#include <hip/hip_runtime.h>
#include <hip/hip_bf16.h>

typedef __attribute__((ext_vector_type(16))) _Float16 v16h;
typedef __attribute__((ext_vector_type(8)))  float    v8f;
typedef int v4i_t __attribute__((ext_vector_type(4)));

#define NB      4
#define DIM     64
#define HPD     32
#define NHEAD   8
#define WSZ     16
#define IMH     128
#define IMW     128
#define NWIN    64                      // windows per image
#define BWIN    (NB*NWIN)               // 256 windows total
#define MM      (WSZ*WSZ)               // 256
#define ATTN_ELEMS ((size_t)BWIN*NHEAD*MM*MM)   // 134,217,728 floats
#define IMG_ELEMS  ((size_t)NB*DIM*IMH*IMW)     // 4,194,304 floats

// CDNA5 async global->LDS copy path (ASYNCcnt). Guarded so compile can't fail
// if the toolchain lacks the builtins.
#if __has_builtin(__builtin_amdgcn_global_load_async_to_lds_b128) && \
    __has_builtin(__builtin_amdgcn_global_load_async_to_lds_b32)
#define USE_ASYNC_LDS 1
#endif

// Typed, address-space-qualified pointer casts matching the builtin signatures
// (AS1 global src, AS3 LDS dst; element type int / v4i per width).
#define GAS32(p)  ((__attribute__((address_space(1))) int*)(p))
#define LAS32(p)  ((__attribute__((address_space(3))) int*)(p))
#define GAS128(p) ((__attribute__((address_space(1))) v4i_t*)(p))
#define LAS128(p) ((__attribute__((address_space(3))) v4i_t*)(p))

__device__ __forceinline__ void wait_asynccnt0() {
#if __has_builtin(__builtin_amdgcn_s_wait_asynccnt)
  __builtin_amdgcn_s_wait_asynccnt(0);
#else
  asm volatile("s_wait_asynccnt 0x0" ::: "memory");
#endif
}

// ---------------------------------------------------------------------------
// Kernel 1: ReflashAttn.  conv(1x3 over N, head-mix) -> conv(1x1)+bias ->
// ReLU -> softmax over N.  One block per (window, m-row).  8 waves; wave w
// owns head w; each lane owns 8 of the 256 n-positions (stride 32).
// Staging uses float4 (b128) loads; halo padded to keep 16B alignment.
// ---------------------------------------------------------------------------
__global__ void reflash_attn_kernel(const float* __restrict__ ra,
                                    const float* __restrict__ w1,
                                    const float* __restrict__ w2,
                                    const float* __restrict__ b2,
                                    float* __restrict__ attn_out) {
  __shared__ float s_in[NHEAD][MM + 8];   // data at offset +4; zeros at [3],[MM+4]
  __shared__ float s_a1[NHEAD][MM];
  __shared__ float s_w1[NHEAD * NHEAD * 3];
  __shared__ float s_w2[NHEAD * NHEAD];
  __shared__ float s_b2[NHEAD];

  const int t  = threadIdx.x;
  const int bw = blockIdx.x >> 8;     // window index
  const int m  = blockIdx.x & 255;    // row within attention matrix

  if (t < NHEAD * NHEAD * 3) s_w1[t] = w1[t];
  if (t < NHEAD * NHEAD)     s_w2[t] = w2[t];
  if (t < NHEAD) { s_b2[t] = b2[t]; s_in[t][3] = 0.f; s_in[t][MM + 4] = 0.f; }

  const size_t base = ((size_t)bw * NHEAD) * MM * MM + (size_t)m * MM;
  // 512 float4 chunks (8 heads x 64), 2 per thread, 16B-aligned on both ends
#pragma unroll
  for (int it = 0; it < 2; ++it) {
    int fi = t + it * 256;
    int hh = fi >> 6, nn4 = fi & 63;
    const float4 vv = *reinterpret_cast<const float4*>(ra + base + (size_t)hh * MM * MM + nn4 * 4);
    *reinterpret_cast<float4*>(&s_in[hh][4 + nn4 * 4]) = vv;
  }
  __syncthreads();

  // conv 1x3 mixing heads (input at offset +4; n-1 -> [n+3], n+1 -> [n+5])
  for (int idx = t; idx < NHEAD * MM; idx += 256) {
    int o = idx >> 8, n = idx & 255;
    float acc = 0.f;
#pragma unroll
    for (int i = 0; i < NHEAD; ++i) {
      const float* wr = &s_w1[(o * NHEAD + i) * 3];
      acc += wr[0] * s_in[i][n + 3] + wr[1] * s_in[i][n + 4] + wr[2] * s_in[i][n + 5];
    }
    s_a1[o][n] = acc;
  }
  __syncthreads();

  // conv 1x1 + bias + ReLU, then softmax over n (per head row)
  const int o = t >> 5;           // head == wave id (8 waves)
  const int lane = t & 31;
  float v[8];
#pragma unroll
  for (int j = 0; j < 8; ++j) {
    int n = lane + 32 * j;
    float acc = s_b2[o];
#pragma unroll
    for (int i = 0; i < NHEAD; ++i) acc += s_w2[o * NHEAD + i] * s_a1[i][n];
    v[j] = fmaxf(acc, 0.f);
  }
  float mx = v[0];
#pragma unroll
  for (int j = 1; j < 8; ++j) mx = fmaxf(mx, v[j]);
  for (int off = 16; off > 0; off >>= 1) mx = fmaxf(mx, __shfl_xor(mx, off));
  float sum = 0.f;
#pragma unroll
  for (int j = 0; j < 8; ++j) { v[j] = __expf(v[j] - mx); sum += v[j]; }
  for (int off = 16; off > 0; off >>= 1) sum += __shfl_xor(sum, off);
  const float inv = 1.f / sum;
#pragma unroll
  for (int j = 0; j < 8; ++j) {
    int n = lane + 32 * j;
    attn_out[base + (size_t)o * MM * MM + n] = v[j] * inv;
  }
}

// ---------------------------------------------------------------------------
// Kernel 2: windowed attention apply via WMMA with a double-buffered async
// pipeline.  out[d,n] = sum_m attn[m,n]*x[d,m] == X(8x256)*A(256x256) per
// (window, head).  One block per (bw,h); 4 waves; wave w owns N-tiles
// 4w..4w+3.  Attn K-slices (32x256 f32, 32 KB) are staged to LDS with CDNA5
// async b128 copies; slice ks+1 is issued before computing slice ks so the
// fetch overlaps fragment build + 4x v_wmma_f32_16x16x32_f16.
// ---------------------------------------------------------------------------
__global__ void attn_apply_wmma_kernel(const float* __restrict__ attn,
                                       const float* __restrict__ lms,
                                       float* __restrict__ refl) {
  __shared__ float s_x32[NHEAD][MM];     // X as f32 (8 KB)
  __shared__ float s_b32[2][32][MM];     // double-buffered attn slice (64 KB)

  const int t  = threadIdx.x;
  const int bh = blockIdx.x;
  const int bw = bh >> 3, h = bh & 7;
  const int img = bw >> 6, wmod = bw & 63;
  const int w1 = wmod >> 3, w2 = wmod & 7;   // interleaved window position

  // stage X = refined_lms gathered through the (interleaved) window partition
  for (int idx = t; idx < NHEAD * MM; idx += 128) {
    int d = idx >> 8, m = idx & 255;
    int r = m >> 4, cc = m & 15;
    int Hg = r * 8 + w1, Wg = cc * 8 + w2;        // '(h w1)','(w w2)' einops
    int ch = h * NHEAD + d;
    const float* gp = &lms[(((size_t)img * DIM + ch) * IMH + Hg) * IMW + Wg];
#ifdef USE_ASYNC_LDS
    __builtin_amdgcn_global_load_async_to_lds_b32(GAS32(gp), LAS32(&s_x32[d][m]), 0, 0);
#else
    s_x32[d][m] = *gp;
#endif
  }

  const int lane  = t & 31;
  const int wave  = t >> 5;
  const int M     = lane & 15;           // A row / B column owned by this lane
  const int baseK = (lane >> 4) * 8;     // lanes 16-31 hold the +8 K-range
  const size_t abase = ((size_t)bw * NHEAD + h) * MM * MM;

  // issue async copies for one 32x256 f32 K-slice (2048 b128 chunks, 16/thread)
  auto stage_slice = [&](int ks, int bsel) {
#pragma unroll
    for (int it = 0; it < 16; ++it) {
      int c = t + it * 128;
      int mm = c >> 6, nn4 = c & 63;
      const float* gp = &attn[abase + (size_t)(ks * 32 + mm) * MM + nn4 * 4];
#ifdef USE_ASYNC_LDS
      __builtin_amdgcn_global_load_async_to_lds_b128(GAS128(gp),
                                                     LAS128(&s_b32[bsel][mm][nn4 * 4]), 0, 0);
#else
      *reinterpret_cast<float4*>(&s_b32[bsel][mm][nn4 * 4]) =
          *reinterpret_cast<const float4*>(gp);
#endif
    }
  };

  v8f acc[4];
#pragma unroll
  for (int i = 0; i < 4; ++i) acc[i] = (v8f){0, 0, 0, 0, 0, 0, 0, 0};

  stage_slice(0, 0);

  for (int ks = 0; ks < 8; ++ks) {
    const int cur = ks & 1;
#ifdef USE_ASYNC_LDS
    wait_asynccnt0();          // this wave's copies for slice ks have landed
#endif
    __syncthreads();           // ... and every other wave's too

    // prefetch next slice into the other buffer; overlaps with compute below.
    // Safe: buffer cur^1 was last read at iteration ks-1, which completed
    // before this barrier in every wave.
    if (ks < 7) stage_slice(ks + 1, cur ^ 1);

    // A fragment: 16-bit 16x32 layout (VGPR0..3 = K 0..7, VGPR4..7 = K 16..23,
    // lanes 16-31 shifted by +8).  Rows 8..15 are zero padding.
    v16h a;
#pragma unroll
    for (int e = 0; e < 16; ++e) {
      int K = baseK + (e < 8 ? e : e + 8);
      a[e] = (M < NHEAD) ? (_Float16)s_x32[M][ks * 32 + K] : (_Float16)0.0f;
    }
#pragma unroll
    for (int nt = 0; nt < 4; ++nt) {
      int ntile = wave * 4 + nt;
      v16h bmat;
#pragma unroll
      for (int e = 0; e < 16; ++e) {
        int K = baseK + (e < 8 ? e : e + 8);
        bmat[e] = (_Float16)s_b32[cur][K][ntile * 16 + M];
      }
      acc[nt] = __builtin_amdgcn_wmma_f32_16x16x32_f16(
          false, a, false, bmat, (short)0, acc[nt], false, false);
    }
  }

  // D layout: lane L (0..15) holds N=L, VGPR v holds M=v (d channel).
  // Lanes 16-31 hold M=8..15 (zero pad rows) -> skip.
  if (lane < 16) {
#pragma unroll
    for (int nt = 0; nt < 4; ++nt) {
      int n = (wave * 4 + nt) * 16 + lane;
      int r = n >> 4, cc = n & 15;
      int Hg = r * 8 + w1, Wg = cc * 8 + w2;
#pragma unroll
      for (int v = 0; v < 8; ++v) {
        int ch = h * NHEAD + v;
        refl[(((size_t)img * DIM + ch) * IMH + Hg) * IMW + Wg] = acc[nt][v];
      }
    }
  }
}

// ---------------------------------------------------------------------------
// Kernel 3: Residual block: depthwise 3x3 -> ReLU -> pointwise 1x1, + skip.
// Block = 64 channels x 4 adjacent-W pixels.
// ---------------------------------------------------------------------------
__global__ void residual_kernel(const float* __restrict__ refl,
                                const float* __restrict__ dww,
                                const float* __restrict__ dwb,
                                const float* __restrict__ pww,
                                const float* __restrict__ pwb,
                                float* __restrict__ res) {
  __shared__ float s_y1[DIM][4];
  const int t  = threadIdx.x;
  const int c  = t >> 2, lp = t & 3;
  const int p  = blockIdx.x * 4 + lp;
  const int img = p >> 14, rem = p & 16383;
  const int y = rem >> 7, x = rem & 127;

  float acc = dwb[c];
#pragma unroll
  for (int ky = 0; ky < 3; ++ky) {
    int yy = y + ky - 1;
#pragma unroll
    for (int kx = 0; kx < 3; ++kx) {
      int xx = x + kx - 1;
      float vv = 0.f;
      if (yy >= 0 && yy < IMH && xx >= 0 && xx < IMW)
        vv = refl[(((size_t)img * DIM + c) * IMH + yy) * IMW + xx];
      acc += dww[c * 9 + ky * 3 + kx] * vv;
    }
  }
  s_y1[c][lp] = fmaxf(acc, 0.f);
  __syncthreads();

  float acc2 = pwb[c];
  for (int ci = 0; ci < DIM; ++ci) acc2 += pww[c * DIM + ci] * s_y1[ci][lp];
  const size_t oidx = (((size_t)img * DIM + c) * IMH + y) * IMW + x;
  res[oidx] = acc2 + refl[oidx];
}

// ---------------------------------------------------------------------------
// Kernel 4: fuse conv 3x3 (96 -> 64 ch) on concat(res, hp) then gate with
// reflashed.  8x8 pixel tile; 96x10x10 patch staged in LDS (38.4 KB).
// Thread = (out-channel o, pixel-group pg); each computes 16 pixels.
// ---------------------------------------------------------------------------
__global__ void fuse_gate_kernel(const float* __restrict__ res,
                                 const float* __restrict__ hp,
                                 const float* __restrict__ fw,
                                 const float* __restrict__ fb,
                                 const float* __restrict__ refl,
                                 float* __restrict__ out) {
  __shared__ float s_cat[96][10][10];
  const int t   = threadIdx.x;
  const int bi  = blockIdx.x;
  const int img = bi >> 8, tile = bi & 255;
  const int ty = tile >> 4, tx = tile & 15;
  const int y0 = ty * 8, x0 = tx * 8;

  for (int idx = t; idx < 96 * 100; idx += 256) {
    int ci = idx / 100;
    int r2 = idx % 100;
    int rr = r2 / 10, cc = r2 % 10;
    int y = y0 + rr - 1, x = x0 + cc - 1;
    float vv = 0.f;
    if (y >= 0 && y < IMH && x >= 0 && x < IMW) {
      if (ci < DIM) vv = res[(((size_t)img * DIM + ci) * IMH + y) * IMW + x];
      else          vv = hp[(((size_t)img * HPD + (ci - DIM)) * IMH + y) * IMW + x];
    }
    s_cat[ci][rr][cc] = vv;
  }
  __syncthreads();

  const int o  = t & 63;
  const int pg = t >> 6;
  float acc[16];
  const float bias = fb[o];
#pragma unroll
  for (int j = 0; j < 16; ++j) acc[j] = bias;

  for (int ci = 0; ci < 96; ++ci) {
#pragma unroll
    for (int ky = 0; ky < 3; ++ky) {
#pragma unroll
      for (int kx = 0; kx < 3; ++kx) {
        float wv = fw[((o * 96 + ci) * 3 + ky) * 3 + kx];
#pragma unroll
        for (int j = 0; j < 16; ++j) {
          int lp = pg * 16 + j;
          int py = lp >> 3, px = lp & 7;
          acc[j] += wv * s_cat[ci][py + ky][px + kx];
        }
      }
    }
  }

#pragma unroll
  for (int j = 0; j < 16; ++j) {
    int lp = pg * 16 + j;
    int py = lp >> 3, px = lp & 7;
    const size_t oidx = (((size_t)img * DIM + o) * IMH + (y0 + py)) * IMW + (x0 + px);
    out[oidx] = refl[oidx] * acc[j];
  }
}

// ---------------------------------------------------------------------------
extern "C" void kernel_launch(void* const* d_in, const int* in_sizes, int n_in,
                              void* d_out, int out_size, void* d_ws, size_t ws_size,
                              hipStream_t stream) {
  (void)in_sizes; (void)n_in; (void)out_size; (void)ws_size;

  const float* ra   = (const float*)d_in[0];   // reused_attn (256,8,256,256)
  const float* lms  = (const float*)d_in[1];   // refined_lms (4,64,128,128)
  const float* hp   = (const float*)d_in[2];   // hp_in       (4,32,128,128)
  const float* raw1 = (const float*)d_in[3];   // ra_w1 (8,8,1,3)
  const float* raw2 = (const float*)d_in[4];   // ra_w2 (8,8,1,1)
  const float* rab2 = (const float*)d_in[5];   // ra_b2 (8)
  const float* dww  = (const float*)d_in[6];   // dw_w  (64,1,3,3)
  const float* dwb  = (const float*)d_in[7];   // dw_b  (64)
  const float* pww  = (const float*)d_in[8];   // pw_w  (64,64,1,1)
  const float* pwb  = (const float*)d_in[9];   // pw_b  (64)
  const float* fw   = (const float*)d_in[10];  // fuse_w (64,96,3,3)
  const float* fb   = (const float*)d_in[11];  // fuse_b (64)

  float* attn_out  = (float*)d_out;            // first tuple element
  float* final_out = attn_out + ATTN_ELEMS;    // second tuple element
  float* refl = (float*)d_ws;                  // reflashed (16 MB)
  float* res  = refl + IMG_ELEMS;              // residual output (16 MB)

  reflash_attn_kernel<<<BWIN * MM, 256, 0, stream>>>(ra, raw1, raw2, rab2, attn_out);
  attn_apply_wmma_kernel<<<BWIN * NHEAD, 128, 0, stream>>>(attn_out, lms, refl);
  residual_kernel<<<(NB * IMH * IMW) / 4, 256, 0, stream>>>(refl, dww, dwb, pww, pwb, res);
  fuse_gate_kernel<<<NB * 256, 256, 0, stream>>>(res, hp, fw, fb, refl, final_out);
}
